// R3DAttention_32495722562135
// MI455X (gfx1250) — matrix operand order repaired and checked by reading, not compiled, not run on hardware
//
#include <hip/hip_runtime.h>

// ---------------------------------------------------------------------------
// MHA forward for MI455X (gfx1250, wave32, WMMA 16x16x32 f16 -> f32)
// Double-buffered LDS pipelines, CDNA5 async global->LDS copies, and
// sched_barrier-pinned prefetch so next-tile VMEM overlaps the WMMA burst.
// ---------------------------------------------------------------------------
typedef _Float16 h16;
typedef __attribute__((ext_vector_type(4)))  float    v4f;
typedef __attribute__((ext_vector_type(8)))  float    v8f;
typedef __attribute__((ext_vector_type(8)))  _Float16 v8h;
typedef __attribute__((ext_vector_type(16))) _Float16 v16h;

#define HIDDEN 1024
#define NQKV   3072
#define SEQ    512
#define BATCH  32
#define HEADS  16
#define HDIM   64

#if __has_builtin(__builtin_amdgcn_sched_barrier)
#define SCHED_FENCE() __builtin_amdgcn_sched_barrier(0)
#else
#define SCHED_FENCE()
#endif

// ---- WMMA wrapper: D = A(16x32 f16) * B(32x16 f16) + C(16x16 f32) ----------
__device__ __forceinline__ v8f wmma_f16(v16h a, v16h b, v8f c) {
  return __builtin_amdgcn_wmma_f32_16x16x32_f16(
      /*neg_a=*/false, a, /*neg_b=*/false, b,
      /*c_mod=*/(short)0, c, /*reuse_a=*/false, /*reuse_b=*/false);
}

// ---- CDNA5 async global->LDS copy (ASYNCcnt-tracked, no VGPR round trip) ---
__device__ __forceinline__ void async_copy_b128(void* lds_dst, const void* gsrc) {
  unsigned loff = (unsigned)(unsigned long long)(uintptr_t)lds_dst;
  unsigned long long g = (unsigned long long)(uintptr_t)gsrc;
  asm volatile("global_load_async_to_lds_b128 %0, %1, off"
               :: "v"(loff), "v"(g) : "memory");
}
__device__ __forceinline__ void wait_async() {
  asm volatile("s_wait_asynccnt 0" ::: "memory");
}

// A-fragment (16x32, f16) from LDS tile stored [row][k], k contiguous.
__device__ __forceinline__ v16h load_a_frag(const h16* lds, int ldk, int row,
                                            int kbase, int lane) {
  int hi = lane >> 4;
  const h16* p = lds + row * ldk + kbase + hi * 8;
  v8h lo = *(const v8h*)p;
  v8h hv = *(const v8h*)(p + 16);
  return __builtin_shufflevector(lo, hv, 0, 1, 2, 3, 4, 5, 6, 7,
                                 8, 9, 10, 11, 12, 13, 14, 15);
}

// B-fragment (32x16, f16) from LDS tile stored TRANSPOSED [n][k].
__device__ __forceinline__ v16h load_b_frag(const h16* lds, int ldk, int nrow,
                                            int kbase, int lane) {
  int hi = lane >> 4;
  const h16* p = lds + nrow * ldk + kbase + hi * 8;
  v8h lo = *(const v8h*)p;
  v8h hv = *(const v8h*)(p + 16);
  return __builtin_shufflevector(lo, hv, 0, 1, 2, 3, 4, 5, 6, 7,
                                 8, 9, 10, 11, 12, 13, 14, 15);
}

__device__ __forceinline__ float rmax16(float v) {
#pragma unroll
  for (int i = 0; i < 4; ++i) v = fmaxf(v, __shfl_xor(v, 1 << i, 32));
  return v;
}
__device__ __forceinline__ float rsum16(float v) {
#pragma unroll
  for (int i = 0; i < 4; ++i) v += __shfl_xor(v, 1 << i, 32);
  return v;
}

// ---------------------------------------------------------------------------
// Kernel 1: QKV = x @ w_qkv + b_qkv (f32 in, f16 LDS tiles, f32 accumulate)
// 128x128 block tile, BK=32, 8 waves (2x4), wave tile 64x32.
// ---------------------------------------------------------------------------
__device__ __forceinline__ void qkv_ldg(const float* __restrict__ x,
                                        const float* __restrict__ w, int m0,
                                        int n0, int kk, int tid, v4f (&ar)[4],
                                        v4f (&br)[4]) {
#pragma unroll
  for (int rep = 0; rep < 4; ++rep) {
    int linear = rep * 256 + tid;            // 0..1023
    int row = linear >> 3, kg = (linear & 7) << 2;
    ar[rep] = *(const v4f*)(x + (size_t)(m0 + row) * HIDDEN + kk + kg);
    int krow = linear >> 5, ng = (linear & 31) << 2;
    br[rep] = *(const v4f*)(w + (size_t)(kk + krow) * NQKV + n0 + ng);
  }
}
__device__ __forceinline__ void qkv_sts(int tid, const v4f (&ar)[4],
                                        const v4f (&br)[4], h16* As, h16* Bs) {
#pragma unroll
  for (int rep = 0; rep < 4; ++rep) {
    int linear = rep * 256 + tid;
    int row = linear >> 3, kg = (linear & 7) << 2;
    h16* dst = As + row * 32 + kg;
    dst[0] = (h16)ar[rep][0]; dst[1] = (h16)ar[rep][1];
    dst[2] = (h16)ar[rep][2]; dst[3] = (h16)ar[rep][3];
    int krow = linear >> 5, ng = (linear & 31) << 2;
#pragma unroll
    for (int j = 0; j < 4; ++j) Bs[(ng + j) * 32 + krow] = (h16)br[rep][j];
  }
}

__global__ __launch_bounds__(256) void mha_qkv_gemm(
    const float* __restrict__ x, const float* __restrict__ w,
    const float* __restrict__ bias, h16* __restrict__ Qh,
    h16* __restrict__ Kh, h16* __restrict__ Vh) {
  __shared__ __align__(16) h16 As[2][128 * 32];  // [m][k]
  __shared__ __align__(16) h16 Bs[2][128 * 32];  // [n][k] transposed

  const int tid = threadIdx.x;
  const int lane = tid & 31, wave = tid >> 5;
  const int hi = lane >> 4, ln = lane & 15;
  const int wm = wave & 1, wn = wave >> 1;
  const int m0 = blockIdx.y * 128;
  const int n0 = blockIdx.x * 128;

  v8f acc[4][2] = {};

  {  // prologue: stage tile 0
    v4f ar[4], br[4];
    qkv_ldg(x, w, m0, n0, 0, tid, ar, br);
    qkv_sts(tid, ar, br, As[0], Bs[0]);
  }
  __syncthreads();

#pragma unroll 2
  for (int it = 0; it < HIDDEN / 32; ++it) {
    const int cur = it & 1;
    v4f ar[4], br[4];
    if (it + 1 < HIDDEN / 32)  // issue next tile's global loads (no wait yet)
      qkv_ldg(x, w, m0, n0, (it + 1) * 32, tid, ar, br);
    SCHED_FENCE();  // pin loads above the WMMA burst

    v16h bf[2], af[4];
#pragma unroll
    for (int nt = 0; nt < 2; ++nt)
      bf[nt] = load_b_frag(Bs[cur], 32, wn * 32 + nt * 16 + ln, 0, lane);
#pragma unroll
    for (int mt = 0; mt < 4; ++mt)
      af[mt] = load_a_frag(As[cur], 32, wm * 64 + mt * 16 + ln, 0, lane);
#pragma unroll
    for (int mt = 0; mt < 4; ++mt)
#pragma unroll
      for (int nt = 0; nt < 2; ++nt)
        acc[mt][nt] = wmma_f16(af[mt], bf[nt], acc[mt][nt]);

    SCHED_FENCE();  // keep cvt+LDS stores (and their loadcnt wait) below
    if (it + 1 < HIDDEN / 32)
      qkv_sts(tid, ar, br, As[cur ^ 1], Bs[cur ^ 1]);
    __syncthreads();
  }

  // Epilogue: + bias, scatter to Q/K/V f16 [b][h][p][d]
#pragma unroll
  for (int nt = 0; nt < 2; ++nt) {
    int n = n0 + wn * 32 + nt * 16 + ln;
    float bv = bias[n];
    int which = n >> 10;
    int c = n & 1023;
    int hh = c >> 6, d = c & 63;
    h16* base = (which == 0) ? Qh : ((which == 1) ? Kh : Vh);
#pragma unroll
    for (int mt = 0; mt < 4; ++mt) {
#pragma unroll
      for (int r = 0; r < 8; ++r) {
        int m = m0 + wm * 64 + mt * 16 + r + 8 * hi;
        int b = m >> 9, p = m & 511;
        base[(size_t)(((b * HEADS + hh) * SEQ) + p) * HDIM + d] =
            (h16)(acc[mt][nt][r] + bv);
      }
    }
  }
}

// ---------------------------------------------------------------------------
// Kernel 2: flash attention per (b,h). Workgroup = 128 q rows, wave = 16 rows.
// K tiles async global->LDS; V register-staged + transposed. Double-buffered.
// ---------------------------------------------------------------------------
__global__ __launch_bounds__(256) void mha_flash_attn(
    const h16* __restrict__ Qh, const h16* __restrict__ Kh,
    const h16* __restrict__ Vh, h16* __restrict__ Zh) {
  __shared__ __align__(16) h16 Qs[128 * 64];       // 16 KB
  __shared__ __align__(16) h16 Ks[2][32 * 64];     // 8 KB  [key][d]
  __shared__ __align__(16) h16 Vs[2][64 * 32];     // 8 KB  [d][key]
  __shared__ __align__(16) h16 Ps[8][16 * 32];     // 8 KB  per-wave P slice

  const int tid = threadIdx.x;
  const int lane = tid & 31, wave = tid >> 5;
  const int hi = lane >> 4, ln = lane & 15;
  const int qb = blockIdx.x, h = blockIdx.y, b = blockIdx.z;

  const size_t bh = ((size_t)b * HEADS + h) * SEQ * HDIM;
  const h16* Qg = Qh + bh + (size_t)qb * 128 * HDIM;
  const h16* Kg = Kh + bh;
  const h16* Vg = Vh + bh;

  const int vkey = tid >> 3;            // V staging: key row 0..31
  const int vd0 = (tid & 7) * 8;        //            d chunk 0..56

  // Stage Q block (async) + chunk 0 of K (async) / V (regs->transposed LDS)
#pragma unroll
  for (int rep = 0; rep < 4; ++rep) {
    int linear = rep * 256 + tid;  // 1024 x 16B = 8192 halves
    async_copy_b128(Qs + linear * 8, Qg + linear * 8);
  }
  async_copy_b128(Ks[0] + tid * 8, Kg + tid * 8);
  {
    v8h vv = *(const v8h*)(Vg + (size_t)vkey * HDIM + vd0);
#pragma unroll
    for (int i = 0; i < 8; ++i) Vs[0][(vd0 + i) * 32 + vkey] = vv[i];
  }
  wait_async();
  __syncthreads();

  v16h aq0 = load_a_frag(Qs, HDIM, wave * 16 + ln, 0, lane);
  v16h aq1 = load_a_frag(Qs, HDIM, wave * 16 + ln, 32, lane);

  v8f o[4] = {};
  float mrow[8], lrow[8];
#pragma unroll
  for (int r = 0; r < 8; ++r) { mrow[r] = -1e30f; lrow[r] = 0.f; }

#pragma unroll 2
  for (int c = 0; c < SEQ / 32; ++c) {
    const int cur = c & 1;
    const int kk = c * 32;
    v8h vreg;
    if (c + 1 < SEQ / 32) {  // issue next chunk: async K, global V -> regs
      async_copy_b128(Ks[cur ^ 1] + tid * 8,
                      Kg + (size_t)(kk + 32) * HDIM + tid * 8);
      vreg = *(const v8h*)(Vg + (size_t)(kk + 32 + vkey) * HDIM + vd0);
    }
    SCHED_FENCE();

    // S = (Q K^T) * 1/sqrt(64) -> two 16x16 tiles; batch B-frag loads first.
    v16h bk[4];
#pragma unroll
    for (int j = 0; j < 2; ++j) {
      bk[2 * j + 0] = load_b_frag(Ks[cur], HDIM, j * 16 + ln, 0, lane);
      bk[2 * j + 1] = load_b_frag(Ks[cur], HDIM, j * 16 + ln, 32, lane);
    }
    v8f s[2];
#pragma unroll
    for (int j = 0; j < 2; ++j) {
      v8f cc = {};
      cc = wmma_f16(aq0, bk[2 * j + 0], cc);
      cc = wmma_f16(aq1, bk[2 * j + 1], cc);
      s[j] = cc * 0.125f;
    }

    // Online softmax; row m = r + 8*hi lives in VGPR r across a 16-lane group.
    float scl[8];
#pragma unroll
    for (int r = 0; r < 8; ++r) {
      float v = rmax16(fmaxf(s[0][r], s[1][r]));
      float nm = fmaxf(mrow[r], v);
      float sc = __expf(mrow[r] - nm);
      mrow[r] = nm;
      float p0 = __expf(s[0][r] - nm);
      float p1 = __expf(s[1][r] - nm);
      s[0][r] = p0; s[1][r] = p1;
      lrow[r] = lrow[r] * sc + rsum16(p0 + p1);
      scl[r] = sc;
    }
#pragma unroll
    for (int t = 0; t < 4; ++t)
#pragma unroll
      for (int r = 0; r < 8; ++r) o[t][r] *= scl[r];

    // C-layout -> A-layout for P via per-wave LDS slice (wave-private).
    h16* P = Ps[wave];
#pragma unroll
    for (int j = 0; j < 2; ++j)
#pragma unroll
      for (int r = 0; r < 8; ++r)
        P[(r + 8 * hi) * 32 + j * 16 + ln] = (h16)s[j][r];
    asm volatile("s_wait_dscnt 0" ::: "memory");  // LDS write->read, same wave

    v16h ap = load_a_frag(P, 32, ln, 0, lane);
    v16h vb[4];
#pragma unroll
    for (int t = 0; t < 4; ++t)
      vb[t] = load_b_frag(Vs[cur], 32, t * 16 + ln, 0, lane);
#pragma unroll
    for (int t = 0; t < 4; ++t) o[t] = wmma_f16(ap, vb[t], o[t]);

    SCHED_FENCE();
    if (c + 1 < SEQ / 32) {  // transpose-store next V chunk after compute
#pragma unroll
      for (int i = 0; i < 8; ++i) Vs[cur ^ 1][(vd0 + i) * 32 + vkey] = vreg[i];
    }
    wait_async();
    __syncthreads();
  }

  // O /= l, store Z f16 at [b][p][h*64+d]
  const int q0 = qb * 128 + wave * 16;
#pragma unroll
  for (int t = 0; t < 4; ++t) {
    int d = t * 16 + ln;
#pragma unroll
    for (int r = 0; r < 8; ++r) {
      int q = q0 + r + 8 * hi;
      Zh[((size_t)(b * SEQ + q)) * HIDDEN + h * HDIM + d] =
          (h16)(o[t][r] * (1.0f / lrow[r]));
    }
  }
}

// ---------------------------------------------------------------------------
// Kernel 3: out = Z @ w_out + b_out. A (f16 Z) async->LDS; B register-staged.
// ---------------------------------------------------------------------------
__device__ __forceinline__ void out_async_a(const h16* __restrict__ Zh, int m0,
                                            int kk, int tid, h16* As) {
#pragma unroll
  for (int rep = 0; rep < 2; ++rep) {
    int linear = rep * 256 + tid;  // 0..511
    int row = linear >> 2, kg = (linear & 3) * 8;
    async_copy_b128(As + row * 32 + kg,
                    Zh + (size_t)(m0 + row) * HIDDEN + kk + kg);
  }
}
__device__ __forceinline__ void out_ldg_b(const float* __restrict__ w, int n0,
                                          int kk, int tid, v4f (&br)[4]) {
#pragma unroll
  for (int rep = 0; rep < 4; ++rep) {
    int linear = rep * 256 + tid;
    int krow = linear >> 5, ng = (linear & 31) << 2;
    br[rep] = *(const v4f*)(w + (size_t)(kk + krow) * HIDDEN + n0 + ng);
  }
}
__device__ __forceinline__ void out_sts_b(int tid, const v4f (&br)[4], h16* Bs) {
#pragma unroll
  for (int rep = 0; rep < 4; ++rep) {
    int linear = rep * 256 + tid;
    int krow = linear >> 5, ng = (linear & 31) << 2;
#pragma unroll
    for (int j = 0; j < 4; ++j) Bs[(ng + j) * 32 + krow] = (h16)br[rep][j];
  }
}

__global__ __launch_bounds__(256) void mha_out_gemm(
    const h16* __restrict__ Zh, const float* __restrict__ w,
    const float* __restrict__ bias, float* __restrict__ out) {
  __shared__ __align__(16) h16 As[2][128 * 32];
  __shared__ __align__(16) h16 Bs[2][128 * 32];

  const int tid = threadIdx.x;
  const int lane = tid & 31, wave = tid >> 5;
  const int hi = lane >> 4, ln = lane & 15;
  const int wm = wave & 1, wn = wave >> 1;
  const int m0 = blockIdx.y * 128;
  const int n0 = blockIdx.x * 128;

  v8f acc[4][2] = {};

  {  // prologue
    out_async_a(Zh, m0, 0, tid, As[0]);
    v4f br[4];
    out_ldg_b(w, n0, 0, tid, br);
    out_sts_b(tid, br, Bs[0]);
    wait_async();
  }
  __syncthreads();

#pragma unroll 2
  for (int it = 0; it < HIDDEN / 32; ++it) {
    const int cur = it & 1;
    v4f br[4];
    if (it + 1 < HIDDEN / 32) {
      out_async_a(Zh, m0, (it + 1) * 32, tid, As[cur ^ 1]);
      out_ldg_b(w, n0, (it + 1) * 32, tid, br);
    }
    SCHED_FENCE();

    v16h bf[2], af[4];
#pragma unroll
    for (int nt = 0; nt < 2; ++nt)
      bf[nt] = load_b_frag(Bs[cur], 32, wn * 32 + nt * 16 + ln, 0, lane);
#pragma unroll
    for (int mt = 0; mt < 4; ++mt)
      af[mt] = load_a_frag(As[cur], 32, wm * 64 + mt * 16 + ln, 0, lane);
#pragma unroll
    for (int mt = 0; mt < 4; ++mt)
#pragma unroll
      for (int nt = 0; nt < 2; ++nt)
        acc[mt][nt] = wmma_f16(af[mt], bf[nt], acc[mt][nt]);

    SCHED_FENCE();
    if (it + 1 < HIDDEN / 32) out_sts_b(tid, br, Bs[cur ^ 1]);
    wait_async();
    __syncthreads();
  }

#pragma unroll
  for (int nt = 0; nt < 2; ++nt) {
    int n = n0 + wn * 32 + nt * 16 + ln;
    float bv = bias[n];
#pragma unroll
    for (int mt = 0; mt < 4; ++mt) {
#pragma unroll
      for (int r = 0; r < 8; ++r) {
        int m = m0 + wm * 64 + mt * 16 + r + 8 * hi;
        out[(size_t)m * HIDDEN + n] = acc[mt][nt][r] + bv;
      }
    }
  }
}

// ---------------------------------------------------------------------------
extern "C" void kernel_launch(void* const* d_in, const int* in_sizes, int n_in,
                              void* d_out, int out_size, void* d_ws,
                              size_t ws_size, hipStream_t stream) {
  (void)in_sizes; (void)n_in; (void)out_size; (void)ws_size;
  const float* x     = (const float*)d_in[0];
  const float* w_qkv = (const float*)d_in[1];
  const float* b_qkv = (const float*)d_in[2];
  const float* w_out = (const float*)d_in[3];
  const float* b_out = (const float*)d_in[4];
  float* out = (float*)d_out;

  // Workspace: Q,K,V,Z f16 -> 4 x 32 MB = 128 MB
  char* ws = (char*)d_ws;
  const size_t TENS = (size_t)BATCH * HEADS * SEQ * HDIM;
  h16* Qh = (h16*)(ws);
  h16* Kh = (h16*)(ws + TENS * sizeof(h16) * 1);
  h16* Vh = (h16*)(ws + TENS * sizeof(h16) * 2);
  h16* Zh = (h16*)(ws + TENS * sizeof(h16) * 3);

  mha_qkv_gemm<<<dim3(NQKV / 128, (BATCH * SEQ) / 128), 256, 0, stream>>>(
      x, w_qkv, b_qkv, Qh, Kh, Vh);
  mha_flash_attn<<<dim3(SEQ / 128, HEADS, BATCH), 256, 0, stream>>>(
      Qh, Kh, Vh, Zh);
  mha_out_gemm<<<dim3(HIDDEN / 128, (BATCH * SEQ) / 128), 256, 0, stream>>>(
      Zh, w_out, b_out, out);
}
